// GNNPerAgentModel_32152125178257
// MI455X (gfx1250) — matrix operand order repaired and checked by reading, not compiled, not run on hardware
//
#include <hip/hip_runtime.h>
#include <hip/hip_bf16.h>

// ---------------------------------------------------------------------------
// CDNA5 (gfx1250) implementation: GNN per-agent model forward pass.
// All dense math (3 convs as implicit GEMM, Wp projection, GAT Wg, head Wh1)
// runs on v_wmma_f32_16x16x32_f16. Weights are pre-swizzled into the wave32
// WMMA B-operand register layout so compute waves use contiguous vector loads.
// Dense GEMMs process two M-tiles per wave (shared B fragment, two independent
// accumulators to cover the WMMA->WMMA RAW hazard window).
// ---------------------------------------------------------------------------

typedef __attribute__((ext_vector_type(16))) _Float16 v16h;
typedef __attribute__((ext_vector_type(8)))  _Float16 v8h;
typedef __attribute__((ext_vector_type(8)))  float    v8f;

#define WMMA_F16(A, Bf, C) \
    __builtin_amdgcn_wmma_f32_16x16x32_f16(false, (A), false, (Bf), (short)0, (C), false, false)

// ---------------- problem constants ----------------
static constexpr int BATCH = 512;
static constexpr int NNODE = 16;

// ---------------- workspace layout (bytes, all 256B aligned) ----------------
static constexpr size_t OFF_W1SW  = 0;                       // 6*2*512*2      = 12,288
static constexpr size_t OFF_W2SW  = 12288;                   // 16*4*512*2     = 65,536
static constexpr size_t OFF_W3SW  = 77824;                   // 18*4*512*2     = 73,728
static constexpr size_t OFF_WGSW  = 151552;                  // 9*16*512*2     = 147,456
static constexpr size_t OFF_WH1SW = 299008;                  // 10*16*512*2    = 163,840
static constexpr size_t OFF_WPSW  = 462848;                  // 512*16*512*2   = 8,388,608
static constexpr size_t OFF_OUT1  = 8851456;                 // 512*1521*32*2  = 49,840,128
// region reuse: out1 is dead after conv2, so the following live inside it
static constexpr size_t OFF_OUT3  = 8851456;                 // 512*64*256*2   = 16,777,216
static constexpr size_t OFF_CAMF  = 25628672;                // 512*256*2      = 262,144
static constexpr size_t OFF_NODE  = 25890816;                // 8192*288*2     = 4,718,592
static constexpr size_t OFF_HN    = 30609408;                // 8192*256*4     = 8,388,608
static constexpr size_t OFF_H     = 38998016;                // 8192*256*4     = 8,388,608
static constexpr size_t OFF_AF    = 47386624;                // 512*320*2      = 327,680
static constexpr size_t OFF_Z     = 47714304;                // 512*256*4      = 524,288
static constexpr size_t OFF_OUT2  = 58691584;                // 512*324*64*2   = 21,233,664
static constexpr size_t WS_NEEDED = 79925248;                // ~76.2 MiB

// ---------------------------------------------------------------------------
// Weight swizzle: conv weights [N][CI][KH][KW] (OIHW) -> WMMA B fragments.
// B fragment layout (wave32, f16 16x16x32): lane L holds column n = nt*16+L%16,
// halves e=0..15 hold k = s*32 + (L<16?0:16) + e.
// ---------------------------------------------------------------------------
__global__ void swizzle_convw_kernel(const float* __restrict__ W, _Float16* __restrict__ dst,
                                     int CI, int KH, int KW, int ksteps, int ntiles)
{
    const int idx = blockIdx.x * blockDim.x + threadIdx.x;
    const int total = ksteps * ntiles * 512;
    if (idx >= total) return;
    const int e  = idx & 15;
    const int L  = (idx >> 4) & 31;
    const int nt = (idx >> 9) % ntiles;
    const int s  = idx / (512 * ntiles);
    const int k  = s * 32 + ((L < 16) ? 0 : 16) + e;      // k = (kh*KW+kw)*CI + ci
    const int n  = nt * 16 + (L & 15);
    const int ci  = k % CI;
    const int kwh = k / CI;
    const int kw  = kwh % KW;
    const int kh  = kwh / KW;
    dst[idx] = (_Float16)W[(((size_t)n * CI + ci) * KH + kh) * KW + kw];
}

// Dense [K][N] row-major f32 -> WMMA B fragments (zero-fill k >= K for padding).
__global__ void swizzle_mat_kernel(const float* __restrict__ W, _Float16* __restrict__ dst,
                                   int K, int Nn, int ksteps, int ntiles)
{
    const int idx = blockIdx.x * blockDim.x + threadIdx.x;
    const int total = ksteps * ntiles * 512;
    if (idx >= total) return;
    const int e  = idx & 15;
    const int L  = (idx >> 4) & 31;
    const int nt = (idx >> 9) % ntiles;
    const int s  = idx / (512 * ntiles);
    const int k  = s * 32 + ((L < 16) ? 0 : 16) + e;
    const int n  = nt * 16 + (L & 15);
    const float v = (k < K) ? W[(size_t)k * Nn + n] : 0.f;
    dst[idx] = (_Float16)v;
}

// ---------------------------------------------------------------------------
// Implicit-GEMM conv with WMMA. One wave computes a 16(M) x OC output tile.
// im2col order k = (kh*KW + kw)*IC + ci keeps 8-consecutive-k groups
// contiguous in memory (IC*KW is a multiple of 8 for all three convs).
// A fragment per lane = two contiguous 8-half chunks per k-step; the A
// fragment is shared across all NT B tiles (NT wmma per gather).
// ---------------------------------------------------------------------------
template <int IH, int IW, int IC, int OC, int KSZ, int STR, int OH, int OW,
          bool IN_F32, bool CMAJOR_OUT>
__global__ __launch_bounds__(256)
void conv_wmma_kernel(const void* __restrict__ inV, const float* __restrict__ bias,
                      const _Float16* __restrict__ Bsw, _Float16* __restrict__ out)
{
    constexpr int M  = OH * OW;
    constexpr int MT = (M + 15) / 16;
    constexpr int NT = OC / 16;
    constexpr int K  = KSZ * KSZ * IC;
    constexpr int KS = K / 32;
    constexpr int RL = KSZ * IC;                 // k per kh-row

    const int wid = blockIdx.x * (blockDim.x >> 5) + (threadIdx.x >> 5);
    const int b   = wid / MT;
    const int mt  = wid % MT;
    if (b >= BATCH) return;

    const int lane  = threadIdx.x & 31;
    const int lhalf = lane >> 4;
    const int lmod  = lane & 15;
    const int koff  = lhalf * 16;

    int m  = mt * 16 + lmod;
    int mc = (m < M) ? m : (M - 1);
    const int oy = mc / OW, ox = mc % OW;

    const _Float16* Bp = Bsw + (size_t)lane * 16;   // running B pointer
    v8f acc[NT] = {};
    for (int s = 0; s < KS; ++s) {
        union { v16h v; v8h h[2]; _Float16 e[16]; } a;
#pragma unroll
        for (int c = 0; c < 2; ++c) {
            const int k0  = s * 32 + koff + c * 8;       // multiple of 8
            const int kh  = k0 / RL;
            const int rem = k0 - kh * RL;
            const int iy  = oy * STR + kh;
            const size_t base = ((size_t)(b * IH + iy) * IW + ox * STR) * IC + rem;
            if constexpr (IN_F32) {
                const float* ip = (const float*)inV + base;
#pragma unroll
                for (int t = 0; t < 8; ++t) a.e[c * 8 + t] = (_Float16)ip[t];
            } else {
                a.h[c] = *(const v8h*)((const _Float16*)inV + base);
            }
        }
#pragma unroll
        for (int t = 0; t < NT; ++t) {
            const v16h bf = *(const v16h*)(Bp + (size_t)t * 512);
            acc[t] = WMMA_F16(a.v, bf, acc[t]);
        }
        Bp += (size_t)NT * 512;
    }
#pragma unroll
    for (int t = 0; t < NT; ++t) {
        const int   n  = t * 16 + lmod;
        const float bv = bias[n];
#pragma unroll
        for (int r = 0; r < 8; ++r) {
            const int mm = mt * 16 + r + lhalf * 8;
            if (mm < M) {
                float v = acc[t][r] + bv;
                v = v > 0.f ? v : 0.f;                    // ReLU
                if constexpr (CMAJOR_OUT)
                    out[((size_t)b * OC + n) * M + mm] = (_Float16)v;   // [b][c][p]
                else
                    out[((size_t)b * M + mm) * OC + n] = (_Float16)v;   // [b][p][c]
            }
        }
    }
}

// ---------------------------------------------------------------------------
// Generic dense GEMM with WMMA: A [M][Kpad] f16 (K contiguous), pre-swizzled B.
// One wave computes a 32(M) x 16(N) tile: two A fragments share each B
// fragment, two independent accumulators. M must be a multiple of 32.
// ---------------------------------------------------------------------------
__global__ __launch_bounds__(256)
void gemm_wmma_kernel(const _Float16* __restrict__ A, const _Float16* __restrict__ Bsw,
                      const float* __restrict__ bias, float* __restrict__ outF,
                      _Float16* __restrict__ outH,
                      int M, int Kpad, int ksteps, int Nn, int ntiles, int relu)
{
    const int wid    = blockIdx.x * (blockDim.x >> 5) + (threadIdx.x >> 5);
    const int mpairs = M >> 5;
    if (wid >= mpairs * ntiles) return;
    const int mp = wid / ntiles, nt = wid % ntiles;

    const int lane  = threadIdx.x & 31;
    const int lhalf = lane >> 4;
    const int lmod  = lane & 15;
    const int koff  = lhalf * 16;

    const _Float16* Arow0 = A + (size_t)(mp * 32 + lmod) * Kpad + koff;
    const _Float16* Arow1 = Arow0 + (size_t)16 * Kpad;
    const _Float16* Bp    = Bsw + ((size_t)nt * 32 + lane) * 16;   // running B pointer
    const size_t    bstr  = (size_t)ntiles * 512;

    v8f acc0 = {}, acc1 = {};
    for (int s = 0; s < ksteps; ++s) {
        union { v16h v; v8h h[2]; } a0, a1;
        a0.h[0] = *(const v8h*)(Arow0 + s * 32);
        a0.h[1] = *(const v8h*)(Arow0 + s * 32 + 8);
        a1.h[0] = *(const v8h*)(Arow1 + s * 32);
        a1.h[1] = *(const v8h*)(Arow1 + s * 32 + 8);
        const v16h bf = *(const v16h*)Bp;
        Bp += bstr;
        acc0 = WMMA_F16(a0.v, bf, acc0);
        acc1 = WMMA_F16(a1.v, bf, acc1);
    }
    const int   n  = nt * 16 + lmod;
    const float bv = bias ? bias[n] : 0.f;
#pragma unroll
    for (int r = 0; r < 8; ++r) {
        const int mm0 = mp * 32 + r + lhalf * 8;
        const int mm1 = mm0 + 16;
        float v0 = acc0[r] + bv;
        float v1 = acc1[r] + bv;
        if (relu) { v0 = v0 > 0.f ? v0 : 0.f; v1 = v1 > 0.f ? v1 : 0.f; }
        if (outH) {
            outH[(size_t)mm0 * Nn + n] = (_Float16)v0;
            outH[(size_t)mm1 * Nn + n] = (_Float16)v1;
        } else {
            outF[(size_t)mm0 * Nn + n] = v0;
            outF[(size_t)mm1 * Nn + n] = v1;
        }
    }
}

// ---------------------------------------------------------------------------
// Build padded node features [B*N][288] f16 = [cam_feat(256) | rel_pos(3) | 0 pad]
// ---------------------------------------------------------------------------
__global__ void node_feats_kernel(const _Float16* __restrict__ camf, const float* __restrict__ pos,
                                  const float* __restrict__ team, _Float16* __restrict__ node)
{
    const int idx = blockIdx.x * blockDim.x + threadIdx.x;
    if (idx >= BATCH * NNODE * 288) return;
    const int k  = idx % 288;
    const int bn = idx / 288;
    const int b  = bn >> 4;
    _Float16 v;
    if (k < 256)      v = camf[b * 256 + k];
    else if (k < 259) v = (_Float16)(team[(size_t)bn * 3 + (k - 256)] - pos[b * 3 + (k - 256)]);
    else              v = (_Float16)0.f;
    node[idx] = v;
}

// ---------------------------------------------------------------------------
// GAT attention: adjacency, masked leaky-relu softmax, weighted sum, ELU.
// One workgroup (256 threads) per image; all tiny state lives in LDS.
// ---------------------------------------------------------------------------
__global__ __launch_bounds__(256)
void gat_attn_kernel(const float* __restrict__ hN, const float* __restrict__ team,
                     const float* __restrict__ a_src, const float* __restrict__ a_dst,
                     float* __restrict__ hOut)
{
    const int b = blockIdx.x;
    const int t = threadIdx.x;
    __shared__ float sT[16][3];
    __shared__ float sAdj[16][16];
    __shared__ float sS[16][2], sD[16][2];
    __shared__ float sAl[16][2][16];
    __shared__ float sRedS[16][16], sRedD[16][16];

    if (t < 48) sT[t / 3][t % 3] = team[b * 48 + t];
    __syncthreads();
    {
        const int i = t >> 4, j = t & 15;
        const float dx = sT[i][0] - sT[j][0];
        const float dy = sT[i][1] - sT[j][1];
        const float dz = sT[i][2] - sT[j][2];
        const float dist = sqrtf(dx * dx + dy * dy + dz * dz);
        sAdj[i][j] = (dist < 100.f && i != j) ? 1.f : 0.f;
    }
    __syncthreads();
    if (t < 16) {                                       // self-loop for isolated nodes
        float rs = 0.f;
        for (int j = 0; j < 16; ++j) rs += sAdj[t][j];
        if (rs == 0.f) sAdj[t][t] = 1.f;
    }
    __syncthreads();

    const float* hNb = hN + (size_t)b * NNODE * 256;
    {
        const int i = t >> 4, r = t & 15;
        for (int h = 0; h < 2; ++h) {
            float as = 0.f, ad = 0.f;
            for (int d = r; d < 128; d += 16) {
                const float v = hNb[i * 256 + h * 128 + d];
                as += v * a_src[h * 128 + d];
                ad += v * a_dst[h * 128 + d];
            }
            sRedS[i][r] = as; sRedD[i][r] = ad;
            __syncthreads();
            if (r == 0) {
                float s0 = 0.f, d0 = 0.f;
                for (int q = 0; q < 16; ++q) { s0 += sRedS[i][q]; d0 += sRedD[i][q]; }
                sS[i][h] = s0; sD[i][h] = d0;
            }
            __syncthreads();
        }
    }
    if (t < 32) {                                       // masked softmax rows
        const int i = t >> 1, h = t & 1;
        const float si = sS[i][h];
        float ev[16];
        float mx = -1e30f;
        for (int j = 0; j < 16; ++j) {
            float e = si + sD[j][h];
            e = e > 0.f ? e : 0.2f * e;                 // leaky_relu(0.2)
            e = sAdj[i][j] > 0.f ? e : -1e9f;
            ev[j] = e;
            mx = e > mx ? e : mx;
        }
        float sum = 0.f;
        for (int j = 0; j < 16; ++j) { const float x = __expf(ev[j] - mx); ev[j] = x; sum += x; }
        const float inv = 1.f / sum;
        for (int j = 0; j < 16; ++j) sAl[i][h][j] = ev[j] * inv;
    }
    __syncthreads();
    {
        const int i = t >> 4, r = t & 15;
        for (int h = 0; h < 2; ++h)
            for (int d = r; d < 128; d += 16) {
                float acc = 0.f;
                for (int j = 0; j < 16; ++j)
                    acc += sAl[i][h][j] * hNb[j * 256 + h * 128 + d];
                const float o = acc > 0.f ? acc : (__expf(acc) - 1.f);   // ELU
                hOut[(size_t)b * 4096 + i * 256 + h * 128 + d] = o;
            }
    }
}

// ---------------------------------------------------------------------------
// Gather agent node + id embedding -> fused feature [B][320] f16
// ---------------------------------------------------------------------------
__global__ void fuse_kernel(const float* __restrict__ hOut, const int* __restrict__ aidx,
                            const float* __restrict__ Wid, const float* __restrict__ bid,
                            _Float16* __restrict__ af)
{
    const int b = blockIdx.x, t = threadIdx.x;          // blockDim = 320
    const int ai = aidx[b];
    if (t < 256) {
        af[(size_t)b * 320 + t] = (_Float16)hOut[((size_t)b * NNODE + ai) * 256 + t];
    } else {
        const int j = t - 256;
        float v = (float)ai * Wid[j] + bid[j];
        v = v > 0.f ? v : 0.f;
        af[(size_t)b * 320 + t] = (_Float16)v;
    }
}

// ---------------------------------------------------------------------------
// Final tiny head: logits = z @ Wh2 + bh2   ([512,256] x [256,6])
// ---------------------------------------------------------------------------
__global__ void head2_kernel(const float* __restrict__ z, const float* __restrict__ Wh2,
                             const float* __restrict__ bh2, float* __restrict__ out)
{
    const int idx = blockIdx.x * blockDim.x + threadIdx.x;
    if (idx >= BATCH * 6) return;
    const int b = idx / 6, j = idx % 6;
    float acc = bh2[j];
    const float* zr = z + (size_t)b * 256;
    for (int k = 0; k < 256; ++k) acc += zr[k] * Wh2[k * 6 + j];
    out[idx] = acc;
}

// ---------------------------------------------------------------------------
extern "C" void kernel_launch(void* const* d_in, const int* in_sizes, int n_in,
                              void* d_out, int out_size, void* d_ws, size_t ws_size,
                              hipStream_t stream)
{
    if (ws_size < WS_NEEDED) return;   // need ~76.2 MiB scratch

    const float* cam   = (const float*)d_in[0];
    const float* pos   = (const float*)d_in[1];
    const float* team  = (const float*)d_in[2];
    const int*   aidx  = (const int*)d_in[3];
    const float* W1    = (const float*)d_in[4];
    const float* b1    = (const float*)d_in[5];
    const float* W2    = (const float*)d_in[6];
    const float* b2    = (const float*)d_in[7];
    const float* W3    = (const float*)d_in[8];
    const float* b3    = (const float*)d_in[9];
    const float* Wp    = (const float*)d_in[10];
    const float* bp    = (const float*)d_in[11];
    const float* Wid   = (const float*)d_in[12];
    const float* bid   = (const float*)d_in[13];
    const float* Wg    = (const float*)d_in[14];
    const float* a_src = (const float*)d_in[15];
    const float* a_dst = (const float*)d_in[16];
    const float* Wh1   = (const float*)d_in[17];
    const float* bh1   = (const float*)d_in[18];
    const float* Wh2   = (const float*)d_in[19];
    const float* bh2   = (const float*)d_in[20];
    float* out = (float*)d_out;

    char* ws = (char*)d_ws;
    _Float16* W1sw  = (_Float16*)(ws + OFF_W1SW);
    _Float16* W2sw  = (_Float16*)(ws + OFF_W2SW);
    _Float16* W3sw  = (_Float16*)(ws + OFF_W3SW);
    _Float16* Wgsw  = (_Float16*)(ws + OFF_WGSW);
    _Float16* Wh1sw = (_Float16*)(ws + OFF_WH1SW);
    _Float16* Wpsw  = (_Float16*)(ws + OFF_WPSW);
    _Float16* out1  = (_Float16*)(ws + OFF_OUT1);
    _Float16* out2  = (_Float16*)(ws + OFF_OUT2);
    _Float16* out3  = (_Float16*)(ws + OFF_OUT3);
    _Float16* camf  = (_Float16*)(ws + OFF_CAMF);
    _Float16* node  = (_Float16*)(ws + OFF_NODE);
    float*    hN    = (float*)   (ws + OFF_HN);
    float*    hOut  = (float*)   (ws + OFF_H);
    _Float16* af    = (_Float16*)(ws + OFF_AF);
    float*    z     = (float*)   (ws + OFF_Z);

    // --- weight swizzles (B operands in WMMA register layout) ---
    swizzle_convw_kernel<<<24,   256, 0, stream>>>(W1, W1sw,  3, 8, 8,  6,  2);
    swizzle_convw_kernel<<<128,  256, 0, stream>>>(W2, W2sw, 32, 4, 4, 16,  4);
    swizzle_convw_kernel<<<144,  256, 0, stream>>>(W3, W3sw, 64, 3, 3, 18,  4);
    swizzle_mat_kernel <<<16384, 256, 0, stream>>>(Wp,  Wpsw,  16384, 256, 512, 16);
    swizzle_mat_kernel <<<288,   256, 0, stream>>>(Wg,  Wgsw,    259, 256,   9, 16);
    swizzle_mat_kernel <<<320,   256, 0, stream>>>(Wh1, Wh1sw,   320, 256,  10, 16);

    // --- CNN encoder: 3 implicit-GEMM WMMA convs ---
    // conv1: 84x84x3 -> 39x39x32 (stride 2, 8x8), M=1521 -> 96 tiles/image
    conv_wmma_kernel<84,84,3,32,8,2,39,39,true,false>
        <<<6144, 256, 0, stream>>>(cam, b1, W1sw, out1);
    // conv2: 39x39x32 -> 18x18x64 (stride 2, 4x4), M=324 -> 21 tiles/image
    conv_wmma_kernel<39,39,32,64,4,2,18,18,false,false>
        <<<1344, 256, 0, stream>>>(out1, b2, W2sw, out2);
    // conv3: 18x18x64 -> 16x16x64 (stride 1, 3x3), channel-major store for flatten
    conv_wmma_kernel<18,18,64,64,3,1,16,16,false,true>
        <<<1024, 256, 0, stream>>>(out2, b3, W3sw, out3);

    // --- projection: [512,16384] @ Wp -> relu -> cam_feat f16 [512,256] ---
    // 16 m-pairs x 16 n-tiles = 256 waves
    gemm_wmma_kernel<<<32, 256, 0, stream>>>(out3, Wpsw, bp, nullptr, camf,
                                             512, 16384, 512, 256, 16, 1);

    // --- node features + GAT linear hN = node @ Wg (M=8192, K=288 padded) ---
    node_feats_kernel<<<9216, 256, 0, stream>>>(camf, pos, team, node);
    // 256 m-pairs x 16 n-tiles = 4096 waves
    gemm_wmma_kernel<<<512, 256, 0, stream>>>(node, Wgsw, nullptr, hN, nullptr,
                                              8192, 288, 9, 256, 16, 0);

    // --- attention softmax + aggregation + ELU ---
    gat_attn_kernel<<<BATCH, 256, 0, stream>>>(hN, team, a_src, a_dst, hOut);

    // --- fuse agent node + id embedding, policy head ---
    fuse_kernel<<<BATCH, 320, 0, stream>>>(hOut, aidx, Wid, bid, af);
    gemm_wmma_kernel<<<32, 256, 0, stream>>>(af, Wh1sw, bh1, z, nullptr,
                                             512, 320, 10, 256, 16, 1);
    head2_kernel<<<12, 256, 0, stream>>>(z, Wh2, bh2, out);
}
